// spatialAttentionScaledGCN_59313498358063
// MI455X (gfx1250) — matrix operand order repaired and checked by reading, not compiled, not run on hardware
//
#include <hip/hip_runtime.h>
#include <hip/hip_bf16.h>
#include <math.h>

// ---------------------------------------------------------------------------
// CDNA5 (gfx1250) implementation: bf16 WMMA 16x16x32, f32 accumulate, wave32.
// ---------------------------------------------------------------------------

typedef __bf16 bf16;
typedef __attribute__((ext_vector_type(8)))  bf16  v8bf;
typedef __attribute__((ext_vector_type(16))) bf16  v16bf;
typedef __attribute__((ext_vector_type(8)))  float v8f;

#define B_   4
#define N_   256
#define T_   30
#define F_   128
#define FO_  128
#define INVSQ 0.08838834764831845f   // 1/sqrt(128)

__device__ __forceinline__ v8f wmma_bf16(v16bf a, v16bf b, v8f c) {
  // (neg_a, A, neg_b, B, c_mod, C, reuse_a, reuse_b)
  return __builtin_amdgcn_wmma_f32_16x16x32_bf16(false, a, false, b, (short)0, c,
                                                 false, false);
}

// A-operand fragment, 16x32 bf16 tile from row-major LDS [rows][ld].
// ISA layout: lanes 0-15 -> M=lane, K = {0..7,16..23}; lanes 16-31 -> same M,
// K = {8..15,24..31}.  Two contiguous 16-byte chunks per lane.
__device__ __forceinline__ v16bf load_a_frag(const bf16* base, int row0, int k0,
                                             int ld) {
  int lane = threadIdx.x & 31;
  int h = lane >> 4, r = lane & 15;
  const bf16* p0 = base + (size_t)(row0 + r) * ld + k0 + 8 * h;
  v8bf lo = *(const v8bf*)p0;
  v8bf hi = *(const v8bf*)(p0 + 16);
  return __builtin_shufflevector(lo, hi, 0, 1, 2, 3, 4, 5, 6, 7, 8, 9, 10, 11,
                                 12, 13, 14, 15);
}

// B-operand fragment, 32x16 bf16 tile: N = lane%16, elements = K in table order
// (K = 8h+s for s<8, 16+8h+(s-8) for s>=8). Source LDS holds B row-major.
__device__ __forceinline__ v16bf load_b_frag(const bf16* base, int k0, int col0,
                                             int ld) {
  int lane = threadIdx.x & 31;
  int h = lane >> 4;
  const bf16* p = base + (size_t)(k0 + 8 * h) * ld + col0 + (lane & 15);
  v16bf v;
  #pragma unroll
  for (int s = 0; s < 8; ++s) {
    v[s] = p[(size_t)s * ld];
    v[s + 8] = p[(size_t)(16 + s) * ld];
  }
  return v;
}

// ---------------------------------------------------------------------------
// Kernel A: per-(b,n) temporal attention gate (30x30, VALU — 6% of FLOPs).
// gate[b][t][n] = softmax_s(X X^T * inv)[t,:] . W_lin
// ---------------------------------------------------------------------------
__global__ __launch_bounds__(128) void gate_kernel(const float* __restrict__ x,
                                                   const float* __restrict__ W_lin,
                                                   float* __restrict__ gate_out) {
  __shared__ float X[T_][F_];
  __shared__ float G[T_][T_];
  int b = blockIdx.x / N_, n = blockIdx.x % N_;
  const float* xg = x + ((size_t)b * N_ + n) * T_ * F_;
  for (int i = threadIdx.x; i < T_ * F_; i += 128) X[i / F_][i % F_] = xg[i];
  __syncthreads();
  for (int p = threadIdx.x; p < T_ * T_; p += 128) {
    int t = p / T_, s = p % T_;
    float d = 0.f;
    #pragma unroll 8
    for (int f = 0; f < F_; ++f) d += X[t][f] * X[s][f];
    G[t][s] = d * INVSQ;
  }
  __syncthreads();
  if (threadIdx.x < T_) {
    int t = threadIdx.x;
    float m = -1e30f;
    for (int s = 0; s < T_; ++s) m = fmaxf(m, G[t][s]);
    float sum = 0.f, g = 0.f;
    for (int s = 0; s < T_; ++s) {
      float e = __expf(G[t][s] - m);
      sum += e;
      g += e * W_lin[s];
    }
    gate_out[((size_t)b * T_ + t) * N_ + n] = g / sum;
  }
}

// ---------------------------------------------------------------------------
// Kernel B: one block per (b, t=1..29).  S = Xp Xn^T (WMMA), row softmax,
// A = phase*softmax*gate, transpose-store to LDS, pred = A^T Xp (WMMA),
// scatter into missing entries of the next x buffer.
// LDS: Xp 64KB + XnT 64KB + At 128KB = 256KB (<= 320KB/WGP).
// ---------------------------------------------------------------------------
__global__ __launch_bounds__(512) void predict_kernel(
    const float* __restrict__ x_cur, float* __restrict__ x_next,
    const float* __restrict__ phase, const float* __restrict__ gate,
    const int* __restrict__ mask) {
  extern __shared__ __align__(128) char smem_raw[];
  bf16* Xp  = (bf16*)smem_raw;      // [N_][F_]  x_prev rows, row-major
  bf16* XnT = Xp + N_ * F_;         // [F_][N_]  x_next transposed
  bf16* At  = XnT + F_ * N_;        // [N_][N_]  At[n][m] = A_{m,n}

  int b = blockIdx.x / (T_ - 1);
  int t = blockIdx.x % (T_ - 1) + 1;
  int tid = threadIdx.x;

  const float* phase_blk = phase + (((size_t)b * T_ + t) * N_) * N_;
  __builtin_prefetch(phase_blk + (size_t)tid * (N_ * N_ / 512), 0, 1);

  // Stage + f32->bf16 convert.
  for (int i = tid; i < N_ * F_; i += 512) {
    int n = i >> 7, f = i & (F_ - 1);
    size_t gbase = ((size_t)b * N_ + n) * T_;
    Xp[i] = (bf16)x_cur[(gbase + (t - 1)) * F_ + f];
    XnT[f * N_ + n] = (bf16)x_cur[(gbase + t) * F_ + f];
  }
  __syncthreads();

  int wave = tid >> 5, lane = tid & 31;
  int h = lane >> 4, lc = lane & 15;
  int row0 = wave * 16;  // 16 waves -> each owns a 16-row strip of S

  // MM1: S[row][col] = Xp[row] . Xn[col], K=128 (4 chunks of 32).
  v16bf aF[4];
  #pragma unroll
  for (int kc = 0; kc < 4; ++kc) aF[kc] = load_a_frag(Xp, row0, kc * 32, F_);
  v8f acc[16];
  #pragma unroll
  for (int j = 0; j < 16; ++j) {
    v8f c = {0.f, 0.f, 0.f, 0.f, 0.f, 0.f, 0.f, 0.f};
    #pragma unroll
    for (int kc = 0; kc < 4; ++kc)
      c = wmma_bf16(aF[kc], load_b_frag(XnT, kc * 32, j * 16, N_), c);
    acc[j] = c;
  }

  // Row softmax (scaled by INVSQ) + phase*gate scaling.
  // C layout: acc[j][r] = S[row0 + 8h + r][16j + lc].
  #pragma unroll
  for (int r = 0; r < 8; ++r) {
    int n = row0 + 8 * h + r;
    float m = -1e30f;
    #pragma unroll
    for (int j = 0; j < 16; ++j) m = fmaxf(m, acc[j][r]);
    for (int off = 1; off < 16; off <<= 1) m = fmaxf(m, __shfl_xor(m, off, 32));
    float sum = 0.f;
    #pragma unroll
    for (int j = 0; j < 16; ++j) {
      float e = __expf((acc[j][r] - m) * INVSQ);
      acc[j][r] = e;
      sum += e;
    }
    for (int off = 1; off < 16; off <<= 1) sum += __shfl_xor(sum, off, 32);
    float scale = gate[((size_t)b * T_ + t) * N_ + n] / sum;
    const float* ph = phase_blk + (size_t)n * N_;
    #pragma unroll
    for (int j = 0; j < 16; ++j) acc[j][r] *= scale * ph[j * 16 + lc];
  }
  // Transposed store: At[m][n] = A_{n,m}; contiguous over r -> one 16B store.
  #pragma unroll
  for (int j = 0; j < 16; ++j) {
    int mcol = j * 16 + lc;
    v8bf pk;
    #pragma unroll
    for (int r = 0; r < 8; ++r) pk[r] = (bf16)acc[j][r];
    *(v8bf*)(At + (size_t)mcol * N_ + row0 + 8 * h) = pk;
  }
  __syncthreads();

  // MM2: pred[n][f] = sum_m At[n][m] * Xp[m][f], K=256 (8 chunks).
  v8f acc2[8];
  v8f zero = {0.f, 0.f, 0.f, 0.f, 0.f, 0.f, 0.f, 0.f};
  #pragma unroll
  for (int jf = 0; jf < 8; ++jf) acc2[jf] = zero;
  for (int kk = 0; kk < 8; ++kk) {
    v16bf a2 = load_a_frag(At, row0, kk * 32, N_);
    #pragma unroll
    for (int jf = 0; jf < 8; ++jf)
      acc2[jf] = wmma_bf16(a2, load_b_frag(Xp, kk * 32, jf * 16, F_), acc2[jf]);
  }

  // Write only the missing entries (observed ones never change).
  #pragma unroll
  for (int r = 0; r < 8; ++r) {
    int n = row0 + 8 * h + r;
    if (mask[n * T_ + t] == 0) {
      float* dst = x_next + (((size_t)b * N_ + n) * T_ + t) * F_;
      #pragma unroll
      for (int jf = 0; jf < 8; ++jf) dst[jf * 16 + lc] = acc2[jf][r];
    }
  }
}

// ---------------------------------------------------------------------------
// Kernel C: final stage per (b,t): sa = softmax(Xt Xt^T * inv) * inv,
// A = phase*adj*sa, agg = A^T Xt, out = relu(agg W_theta^T), transposed store.
// ---------------------------------------------------------------------------
__global__ __launch_bounds__(512) void final_kernel(
    const float* __restrict__ x, const float* __restrict__ phase,
    const float* __restrict__ adj, const float* __restrict__ Wth,
    float* __restrict__ out) {
  extern __shared__ __align__(128) char smem_raw[];
  bf16* Xt  = (bf16*)smem_raw;      // [N_][F_]
  bf16* XtT = Xt + N_ * F_;         // [F_][N_]; reused later as agg [N_][F_]
  bf16* At  = XtT + F_ * N_;        // [N_][N_]; reused later as WtT [F_][FO_]

  int b = blockIdx.x / T_;
  int t = blockIdx.x % T_;
  int tid = threadIdx.x;

  for (int i = tid; i < N_ * F_; i += 512) {
    int n = i >> 7, f = i & (F_ - 1);
    float v = x[(((size_t)b * N_ + n) * T_ + t) * F_ + f];
    Xt[i] = (bf16)v;
    XtT[f * N_ + n] = (bf16)v;
  }
  __syncthreads();

  int wave = tid >> 5, lane = tid & 31;
  int h = lane >> 4, lc = lane & 15;
  int row0 = wave * 16;

  // MM1: S = Xt Xt^T
  v16bf aF[4];
  #pragma unroll
  for (int kc = 0; kc < 4; ++kc) aF[kc] = load_a_frag(Xt, row0, kc * 32, F_);
  v8f acc[16];
  #pragma unroll
  for (int j = 0; j < 16; ++j) {
    v8f c = {0.f, 0.f, 0.f, 0.f, 0.f, 0.f, 0.f, 0.f};
    #pragma unroll
    for (int kc = 0; kc < 4; ++kc)
      c = wmma_bf16(aF[kc], load_b_frag(XtT, kc * 32, j * 16, N_), c);
    acc[j] = c;
  }

  const float* phase_blk = phase + (((size_t)b * T_ + t) * N_) * N_;
  #pragma unroll
  for (int r = 0; r < 8; ++r) {
    int n = row0 + 8 * h + r;
    float m = -1e30f;
    #pragma unroll
    for (int j = 0; j < 16; ++j) m = fmaxf(m, acc[j][r]);
    for (int off = 1; off < 16; off <<= 1) m = fmaxf(m, __shfl_xor(m, off, 32));
    float sum = 0.f;
    #pragma unroll
    for (int j = 0; j < 16; ++j) {
      float e = __expf((acc[j][r] - m) * INVSQ);
      acc[j][r] = e;
      sum += e;
    }
    for (int off = 1; off < 16; off <<= 1) sum += __shfl_xor(sum, off, 32);
    float scale = INVSQ / sum;  // sa = softmax * inv
    const float* ph = phase_blk + (size_t)n * N_;
    const float* aj = adj + (size_t)n * N_;
    #pragma unroll
    for (int j = 0; j < 16; ++j) {
      int mcol = j * 16 + lc;
      acc[j][r] *= scale * ph[mcol] * aj[mcol];
    }
  }
  #pragma unroll
  for (int j = 0; j < 16; ++j) {
    int mcol = j * 16 + lc;
    v8bf pk;
    #pragma unroll
    for (int r = 0; r < 8; ++r) pk[r] = (bf16)acc[j][r];
    *(v8bf*)(At + (size_t)mcol * N_ + row0 + 8 * h) = pk;
  }
  __syncthreads();

  // MM2: agg[n][f] = sum_m At[n][m] * Xt[m][f]
  v8f acc2[8];
  v8f zero = {0.f, 0.f, 0.f, 0.f, 0.f, 0.f, 0.f, 0.f};
  #pragma unroll
  for (int jf = 0; jf < 8; ++jf) acc2[jf] = zero;
  for (int kk = 0; kk < 8; ++kk) {
    v16bf a2 = load_a_frag(At, row0, kk * 32, N_);
    #pragma unroll
    for (int jf = 0; jf < 8; ++jf)
      acc2[jf] = wmma_bf16(a2, load_b_frag(Xt, kk * 32, jf * 16, F_), acc2[jf]);
  }
  __syncthreads();  // all reads of XtT/At done -> safe to repurpose

  // Stage agg (bf16, row-major) and W_theta^T for MM3.
  bf16* agg = XtT;  // [N_][F_]
  bf16* WtT = At;   // [F_][FO_]
  #pragma unroll
  for (int r = 0; r < 8; ++r) {
    int n = row0 + 8 * h + r;
    #pragma unroll
    for (int jf = 0; jf < 8; ++jf) agg[n * F_ + jf * 16 + lc] = (bf16)acc2[jf][r];
  }
  for (int i = tid; i < FO_ * F_; i += 512) {
    int o = i >> 7, f = i & (F_ - 1);
    WtT[f * FO_ + o] = (bf16)Wth[i];  // W_theta is (Fo, F) row-major
  }
  __syncthreads();

  // MM3: D[n][o] = sum_f agg[n][f] * W_theta[o][f], then ReLU.
  v8f acc3[8];
  #pragma unroll
  for (int jo = 0; jo < 8; ++jo) acc3[jo] = zero;
  for (int kc = 0; kc < 4; ++kc) {
    v16bf a3 = load_a_frag(agg, row0, kc * 32, F_);
    #pragma unroll
    for (int jo = 0; jo < 8; ++jo)
      acc3[jo] = wmma_bf16(a3, load_b_frag(WtT, kc * 32, jo * 16, FO_), acc3[jo]);
  }
  #pragma unroll
  for (int r = 0; r < 8; ++r) {
    int n = row0 + 8 * h + r;
    float* dst = out + (((size_t)b * N_ + n) * T_ + t) * FO_;  // (B,N,T,Fo)
    #pragma unroll
    for (int jo = 0; jo < 8; ++jo) dst[jo * 16 + lc] = fmaxf(acc3[jo][r], 0.f);
  }
}

// ---------------------------------------------------------------------------
// Host: double-buffered 29-iteration scan + final stage, all on `stream`.
// Workspace layout: [gate B*T*N f32][xA][xB], ~31.7 MB.
// ---------------------------------------------------------------------------
extern "C" void kernel_launch(void* const* d_in, const int* in_sizes, int n_in,
                              void* d_out, int out_size, void* d_ws,
                              size_t ws_size, hipStream_t stream) {
  (void)in_sizes; (void)n_in; (void)out_size; (void)ws_size;
  const float* x     = (const float*)d_in[0];
  const float* phase = (const float*)d_in[1];
  const float* adj   = (const float*)d_in[2];
  const int*   mask  = (const int*)d_in[3];
  const float* Wth   = (const float*)d_in[4];
  const float* Wlin  = (const float*)d_in[5];
  float* outp = (float*)d_out;

  const size_t xElems = (size_t)B_ * N_ * T_ * F_;
  float* gateBuf = (float*)d_ws;
  float* xA = gateBuf + ((size_t)B_ * T_ * N_ + 256);
  float* xB = xA + xElems;

  // Both buffers start as copies of x; observed entries never change, so each
  // iteration only rewrites missing (mask==0, t>=1) entries of the other buf.
  hipMemcpyAsync(xA, x, xElems * sizeof(float), hipMemcpyDeviceToDevice, stream);
  hipMemcpyAsync(xB, x, xElems * sizeof(float), hipMemcpyDeviceToDevice, stream);

  const size_t smemBytes = (size_t)(N_ * F_ + F_ * N_ + N_ * N_) * sizeof(bf16);

  for (int it = 0; it < T_ - 1; ++it) {
    float* cur = (it & 1) ? xB : xA;
    float* nxt = (it & 1) ? xA : xB;
    gate_kernel<<<B_ * N_, 128, 0, stream>>>(cur, Wlin, gateBuf);
    predict_kernel<<<B_ * (T_ - 1), 512, smemBytes, stream>>>(cur, nxt, phase,
                                                              gateBuf, mask);
  }
  float* xf = ((T_ - 2) & 1) ? xA : xB;  // buffer written by last iteration
  final_kernel<<<B_ * T_, 512, smemBytes, stream>>>(xf, phase, adj, Wth, outp);
}
